// WindowAttention_34634616275467
// MI455X (gfx1250) — matrix operand order repaired and checked by reading, not compiled, not run on hardware
//
#include <hip/hip_runtime.h>
#include <hip/hip_bf16.h>

typedef __attribute__((ext_vector_type(16))) __bf16 v16bf;
typedef __attribute__((ext_vector_type(8)))  float  v8f;
typedef __attribute__((ext_vector_type(4)))  float  v4f;

#define NTOK  49
#define NP    64
#define DIM   256
#define HEADS 8
#define DH    32
#define TD    768
#define NW    64

// LDS strides (elements), tuned against the 64x4B bank model
#define LDA   276   // A / O buffer row stride (bf16): 552B rows, 10-bank lane stride (conflict-free)
#define LQK   40    // q,k row stride: 80B rows, 16B aligned (b128-able), 16 distinct banks/16 lanes
#define LV    68    // vT row stride: 136B rows, 8B aligned (packed b64 stores), 2-way worst case
#define LP    66    // attn row stride: 132B rows, 33-bank lane stride (conflict-free)

#define QK_SCALE 0.17677669529663687f  // 1/sqrt(32)

// LDS element layout (bf16 elements, then byte regions)
#define NA_ELEMS   (NP*LDA)            // 17664
#define NQ_ELEMS   (HEADS*NP*LQK)      // 20480
#define NV_ELEMS   (HEADS*DH*LV)       // 17408
#define NP_ELEMS   (HEADS*NP*LP)       // 33792
#define OFF_A      0
#define OFF_Q      (OFF_A + NA_ELEMS)
#define OFF_K      (OFF_Q + NQ_ELEMS)
#define OFF_V      (OFF_K + NQ_ELEMS)
#define OFF_P      (OFF_V + NV_ELEMS)
#define BF16_TOTAL (OFF_P + NP_ELEMS)          // 109824 bf16 elements
#define OFF_REL_B  (BF16_TOTAL*2)              // byte offset
#define REL_BYTES  2560                        // 49*52 rounded up
#define OFF_BIAS_B (OFF_REL_B + REL_BYTES)
#define BIAS_BYTES (169*8*4)                   // 5408
#define OFF_MASK_B (OFF_BIAS_B + BIAS_BYTES)
#define MASK_BYTES 9616                        // 49*49*4 rounded up
#define SMEM_BYTES (OFF_MASK_B + MASK_BYTES)   // 237232 < 320KB

__device__ __forceinline__ v8f v8f_zero() {
  v8f z;
#pragma unroll
  for (int i = 0; i < 8; ++i) z[i] = 0.0f;
  return z;
}

// 16-bit A fragment (16x32, MxK): lane half picks contiguous K-run 0..15 vs 16..31.
__device__ __forceinline__ v16bf load_a16(const __bf16* p, int half) {
  v16bf a;
  const __bf16* p0 = p + 16 * half;
#pragma unroll
  for (int e = 0; e < 16; ++e) a[e] = p0[e];
  return a;
}

// 16-bit B fragment (32x16, KxN): lane = column N, 16 contiguous K elems, halves split K 0..15/16..31.
__device__ __forceinline__ v16bf load_b16_lds(const __bf16* p, int half) {
  v16bf b;
  const __bf16* q = p + 16 * half;
#pragma unroll
  for (int e = 0; e < 16; ++e) b[e] = q[e];
  return b;
}

// B fragment from a global fp32 weight row (row-major W[j][k], lane owns column j).
__device__ __forceinline__ v16bf load_b16_w(const float* p, int half) {
  v16bf b;
  const float* q = p + 16 * half;
#pragma unroll
  for (int e = 0; e < 16; ++e) b[e] = (__bf16)q[e];
  return b;
}

__device__ __forceinline__ v8f wmma_bf16(v16bf a, v16bf b, v8f c) {
  return __builtin_amdgcn_wmma_f32_16x16x32_bf16(false, a, false, b, (short)0, c, false, false);
}

// CDNA5 async global->LDS copy (ASYNCcnt-tracked). lds_off = AS3 byte offset
// (low 32 bits of a generic LDS pointer), gsrc = 64-bit global address.
__device__ __forceinline__ void async_g2lds_b32(unsigned lds_off, const float* gsrc) {
  asm volatile("global_load_async_to_lds_b32 %0, %1, off"
               :: "v"(lds_off), "v"(gsrc) : "memory");
}
__device__ __forceinline__ void wait_asynccnt0() {
  asm volatile("s_wait_asynccnt 0x0" ::: "memory");
}

struct alignas(8) bf4 { __bf16 v[4]; };

extern "C" __global__ void __launch_bounds__(256, 1)
WindowAttention_34634616275467_kernel(const float* __restrict__ x,
                                      const float* __restrict__ mask,
                                      const int*   __restrict__ rel_index,
                                      const float* __restrict__ qkv_w,
                                      const float* __restrict__ qkv_b,
                                      const float* __restrict__ proj_w,
                                      const float* __restrict__ proj_b,
                                      const float* __restrict__ bias_table,
                                      float* __restrict__ out) {
  extern __shared__ char smem[];
  __bf16* ldsA = (__bf16*)smem;                 // x, later O (NP x DIM, stride LDA)
  __bf16* ldsQ = ldsA + OFF_Q;                  // [head][row][dh]    stride LQK
  __bf16* ldsK = ldsA + OFF_K;                  // [head][row][dh]    stride LQK
  __bf16* ldsV = ldsA + OFF_V;                  // [head][dh][row]    stride LV (transposed V)
  __bf16* ldsP = ldsA + OFF_P;                  // [head][row][col]   stride LP
  unsigned char* ldsRel  = (unsigned char*)(smem + OFF_REL_B);  // [49][52] u8
  float*         ldsBias = (float*)(smem + OFF_BIAS_B);         // [169][8] f32
  float*         ldsMask = (float*)(smem + OFF_MASK_B);         // [49][49] f32

  const int tid  = threadIdx.x;
  const int wave = tid >> 5;
  const int lane = tid & 31;
  const int half = lane >> 4;
  const int ln   = lane & 15;
  const int b    = blockIdx.x;
  const int win  = b & (NW - 1);

  // warm this wave's weight stream into L2 (global_prefetch_b8)
  __builtin_prefetch(qkv_w + (size_t)(wave * 96) * DIM, 0, 1);
  __builtin_prefetch(proj_w + (size_t)(wave * 32) * DIM, 0, 1);

  // ---------- Phase 0a: launch async mask -> LDS copy (overlaps with QKV GEMM) ----------
  {
    const float* mw = mask + (size_t)win * NTOK * NTOK;
    for (int idx = tid; idx < NTOK * NTOK; idx += 256) {
      const unsigned dst = (unsigned)(uintptr_t)(ldsMask + idx);
      async_g2lds_b32(dst, mw + idx);
    }
  }

  // ---------- Phase 0b: stage x (zero-padded to 64 rows, bf16), rel_index (u8), bias_table ----------
  for (int idx = tid; idx < NP * (DIM / 4); idx += 256) {
    const int row = idx >> 6;           // 64 float4-chunks per row
    const int c4  = (idx & 63) * 4;
    float v0 = 0.f, v1 = 0.f, v2 = 0.f, v3 = 0.f;
    if (row < NTOK) {
      const v4f xv = *(const v4f*)(x + ((size_t)b * NTOK + row) * DIM + c4);
      v0 = xv[0]; v1 = xv[1]; v2 = xv[2]; v3 = xv[3];
    }
    __bf16* dst = ldsA + row * LDA + c4;
    dst[0] = (__bf16)v0; dst[1] = (__bf16)v1; dst[2] = (__bf16)v2; dst[3] = (__bf16)v3;
  }
  for (int idx = tid; idx < NTOK * NTOK; idx += 256) {
    const int i = idx / NTOK, j = idx - i * NTOK;
    ldsRel[i * 52 + j] = (unsigned char)rel_index[idx];
  }
  for (int idx = tid; idx < 169 * HEADS; idx += 256) ldsBias[idx] = bias_table[idx];
  __syncthreads();

  // ---------- Phase 1: QKV GEMM (64x768 = 64x256 @ 256x768), wave owns 96 cols ----------
  for (int cg = 0; cg < 3; ++cg) {
    const int j0 = wave * 96 + cg * 32;
    v8f acc[4][2];
#pragma unroll
    for (int ti = 0; ti < 4; ++ti)
#pragma unroll
      for (int t = 0; t < 2; ++t) acc[ti][t] = v8f_zero();

    for (int kk = 0; kk < DIM / 32; ++kk) {
      v16bf bfr[2];
#pragma unroll
      for (int t = 0; t < 2; ++t) {
        const int c = j0 + 16 * t + ln;
        bfr[t] = load_b16_w(qkv_w + (size_t)c * DIM + kk * 32, half);
      }
#pragma unroll
      for (int ti = 0; ti < 4; ++ti) {
        const v16bf afr = load_a16(ldsA + (16 * ti + ln) * LDA + kk * 32, half);
#pragma unroll
        for (int t = 0; t < 2; ++t) acc[ti][t] = wmma_bf16(afr, bfr[t], acc[ti][t]);
      }
    }
    // Scatter to per-head Q (scaled), K, and transposed V in LDS
#pragma unroll
    for (int ti = 0; ti < 4; ++ti) {
#pragma unroll
      for (int t = 0; t < 2; ++t) {
        const int c = j0 + 16 * t + ln;
        const float bias = qkv_b[c];
        if (c < 256) {
          const int h = c >> 5, dj = c & 31;
#pragma unroll
          for (int r = 0; r < 8; ++r) {
            const int row = 16 * ti + r + 8 * half;
            ldsQ[(h * NP + row) * LQK + dj] = (__bf16)((acc[ti][t][r] + bias) * QK_SCALE);
          }
        } else if (c < 512) {
          const int c2 = c - 256, h = c2 >> 5, dj = c2 & 31;
#pragma unroll
          for (int r = 0; r < 8; ++r) {
            const int row = 16 * ti + r + 8 * half;
            ldsK[(h * NP + row) * LQK + dj] = (__bf16)(acc[ti][t][r] + bias);
          }
        } else {
          // transposed V: 8 consecutive rows per lane -> two packed 8-byte LDS stores
          const int c2 = c - 512, h = c2 >> 5, dj = c2 & 31;
          bf4 lo, hi;
#pragma unroll
          for (int r = 0; r < 4; ++r) lo.v[r] = (__bf16)(acc[ti][t][r] + bias);
#pragma unroll
          for (int r = 0; r < 4; ++r) hi.v[r] = (__bf16)(acc[ti][t][4 + r] + bias);
          __bf16* vp = ldsV + (h * DH + dj) * LV + 16 * ti + 8 * half;
          *(bf4*)vp = lo;
          *(bf4*)(vp + 4) = hi;
        }
      }
    }
  }
  // mask async copy must be LDS-visible before softmax (barrier below publishes it)
  wait_asynccnt0();
  __syncthreads();

  // ---------- Phase 2: attention, one head per wave ----------
  {
    const int h = wave;
    const __bf16* qh = ldsQ + h * NP * LQK;
    const __bf16* kh = ldsK + h * NP * LQK;

    // S = Q @ K^T  (64x64, K=32 -> one WMMA per 16x16 tile)
    v8f sacc[4][4];
    {
      v16bf bfr[4];
#pragma unroll
      for (int tj = 0; tj < 4; ++tj)
        bfr[tj] = load_b16_lds(kh + (16 * tj + ln) * LQK, half);
#pragma unroll
      for (int ti = 0; ti < 4; ++ti) {
        const v16bf afr = load_a16(qh + (16 * ti + ln) * LQK, half);
#pragma unroll
        for (int tj = 0; tj < 4; ++tj)
          sacc[ti][tj] = wmma_bf16(afr, bfr[tj], v8f_zero());
      }
    }

    // bias + mask + softmax (rows over 49 valid cols), write P bf16 to LDS
#pragma unroll
    for (int ti = 0; ti < 4; ++ti) {
#pragma unroll
      for (int r = 0; r < 8; ++r) {
        const int i = 16 * ti + r + 8 * half;
        float mx = -3.0e38f;
#pragma unroll
        for (int tj = 0; tj < 4; ++tj) {
          const int j = 16 * tj + ln;
          float v = -3.0e38f;
          if (i < NTOK && j < NTOK) {
            const int bi = ldsRel[i * 52 + j];
            v = sacc[ti][tj][r] + ldsBias[bi * HEADS + h] + ldsMask[i * NTOK + j];
          }
          sacc[ti][tj][r] = v;
          mx = fmaxf(mx, v);
        }
#pragma unroll
        for (int d = 1; d < 16; d <<= 1) mx = fmaxf(mx, __shfl_xor(mx, d, 16));
        float sum = 0.0f;
#pragma unroll
        for (int tj = 0; tj < 4; ++tj) {
          const float p = __expf(sacc[ti][tj][r] - mx);
          sacc[ti][tj][r] = p;
          sum += p;
        }
#pragma unroll
        for (int d = 1; d < 16; d <<= 1) sum += __shfl_xor(sum, d, 16);
        const float inv = (sum > 0.0f) ? (1.0f / sum) : 0.0f;
#pragma unroll
        for (int tj = 0; tj < 4; ++tj) {
          const int j = 16 * tj + ln;
          ldsP[(h * NP + i) * LP + j] = (__bf16)(sacc[ti][tj][r] * inv);
        }
      }
    }

    // O = P @ V  (64x32, K=64 -> 2 WMMA steps), write O into ldsA (x no longer needed)
    v8f oacc[4][2];
#pragma unroll
    for (int ti = 0; ti < 4; ++ti)
#pragma unroll
      for (int t = 0; t < 2; ++t) oacc[ti][t] = v8f_zero();
#pragma unroll
    for (int kk = 0; kk < 2; ++kk) {
      v16bf bfr[2];
#pragma unroll
      for (int t = 0; t < 2; ++t)
        bfr[t] = load_b16_lds(ldsV + (h * DH + 16 * t + ln) * LV + kk * 32, half);
#pragma unroll
      for (int ti = 0; ti < 4; ++ti) {
        const v16bf afr = load_a16(ldsP + (h * NP + 16 * ti + ln) * LP + kk * 32, half);
#pragma unroll
        for (int t = 0; t < 2; ++t) oacc[ti][t] = wmma_bf16(afr, bfr[t], oacc[ti][t]);
      }
    }
#pragma unroll
    for (int ti = 0; ti < 4; ++ti)
#pragma unroll
      for (int t = 0; t < 2; ++t) {
        const int c = h * DH + 16 * t + ln;
#pragma unroll
        for (int r = 0; r < 8; ++r) {
          const int row = 16 * ti + r + 8 * half;
          ldsA[row * LDA + c] = (__bf16)oacc[ti][t][r];
        }
      }
  }
  __syncthreads();

  // ---------- Phase 3: projection (64x256 = 64x256 @ 256x256), wave owns 32 cols ----------
  {
    const int j0 = wave * 32;
    v8f pacc[4][2];
#pragma unroll
    for (int ti = 0; ti < 4; ++ti)
#pragma unroll
      for (int t = 0; t < 2; ++t) pacc[ti][t] = v8f_zero();

    for (int kk = 0; kk < DIM / 32; ++kk) {
      v16bf bfr[2];
#pragma unroll
      for (int t = 0; t < 2; ++t) {
        const int c = j0 + 16 * t + ln;
        bfr[t] = load_b16_w(proj_w + (size_t)c * DIM + kk * 32, half);
      }
#pragma unroll
      for (int ti = 0; ti < 4; ++ti) {
        const v16bf afr = load_a16(ldsA + (16 * ti + ln) * LDA + kk * 32, half);
#pragma unroll
        for (int t = 0; t < 2; ++t) pacc[ti][t] = wmma_bf16(afr, bfr[t], pacc[ti][t]);
      }
    }
#pragma unroll
    for (int ti = 0; ti < 4; ++ti)
#pragma unroll
      for (int t = 0; t < 2; ++t) {
        const int c = j0 + 16 * t + ln;
        const float pb = proj_b[c];
#pragma unroll
        for (int r = 0; r < 8; ++r) {
          const int row = 16 * ti + r + 8 * half;
          if (row < NTOK)
            out[((size_t)b * NTOK + row) * DIM + c] = pacc[ti][t][r] + pb;
        }
      }
  }
}

extern "C" void kernel_launch(void* const* d_in, const int* in_sizes, int n_in,
                              void* d_out, int out_size, void* d_ws, size_t ws_size,
                              hipStream_t stream) {
  (void)in_sizes; (void)n_in; (void)out_size; (void)d_ws; (void)ws_size;
  const float* x          = (const float*)d_in[0];
  const float* mask       = (const float*)d_in[1];
  const int*   rel_index  = (const int*)  d_in[2];
  const float* qkv_w      = (const float*)d_in[3];
  const float* qkv_b      = (const float*)d_in[4];
  const float* proj_w     = (const float*)d_in[5];
  const float* proj_b     = (const float*)d_in[6];
  const float* bias_table = (const float*)d_in[7];
  float* out = (float*)d_out;

  hipFuncSetAttribute((const void*)WindowAttention_34634616275467_kernel,
                      hipFuncAttributeMaxDynamicSharedMemorySize, SMEM_BYTES);
  dim3 grid(4096), block(256);
  WindowAttention_34634616275467_kernel<<<grid, block, SMEM_BYTES, stream>>>(
      x, mask, rel_index, qkv_w, qkv_b, proj_w, proj_b, bias_table, out);
}
